// SpikingNeuralNetwork_30425548324896
// MI455X (gfx1250) — matrix operand order repaired and checked by reading, not compiled, not run on hardware
//
#include <hip/hip_runtime.h>
#include <stdint.h>

// ---------------------------------------------------------------------------
// Problem sizes (fixed by reference)
// ---------------------------------------------------------------------------
static constexpr int Bz = 32, Sz = 1024, Iz = 256, Hz = 512, Oz = 256;
static constexpr int Mz = Bz * Sz;          // 32768 rows of the big GEMMs
static constexpr int NWG = 16;              // workgroups in the scan kernel
static constexpr float DECAY = 0.95f;
static constexpr float REFRAC = 2.0f;

#define DEV static __device__ __forceinline__

typedef __bf16 bf16_t;
typedef bf16_t v16bf __attribute__((ext_vector_type(16)));
typedef bf16_t v8bf  __attribute__((ext_vector_type(8)));
typedef float  v8f   __attribute__((ext_vector_type(8)));
typedef unsigned int v4u __attribute__((ext_vector_type(4)));
typedef int v8i __attribute__((ext_vector_type(8)));
typedef int v4i __attribute__((ext_vector_type(4)));

#if defined(__has_builtin)
#if __has_builtin(__builtin_amdgcn_tensor_load_to_lds)
#define HAVE_TDM 1
#endif
#endif

// ---------------------------------------------------------------------------
// bf16 helpers (round-to-nearest-even split of f32 into hi + residual)
// ---------------------------------------------------------------------------
DEV unsigned short f2bf_rne(float f) {
  unsigned u = __float_as_uint(f);
  unsigned r = u + 0x7FFFu + ((u >> 16) & 1u);
  return (unsigned short)(r >> 16);
}
DEV float bf2f(unsigned short b) { return __uint_as_float(((unsigned)b) << 16); }

// ---------------------------------------------------------------------------
// WMMA fragment loads.
// A-matrix 16x32 bf16: lane<16 -> M=lane, K=0..15;
// lane>=16 -> M=lane-16, K=16..31. Caller passes row_base+sel*16 (contiguous).
// B-matrix 32x16 bf16 (stored n-major/transposed): lane<16 -> N=lane K=0..15,
// lane>=16 -> N=lane-16 K=16..31. Caller passes row_base+sel*16 (contiguous).
// ---------------------------------------------------------------------------
DEV v16bf ld_frag_a(const unsigned short* p) {
  v8bf a = *(const v8bf*)p;
  v8bf b = *(const v8bf*)(p + 8);
  return __builtin_shufflevector(a, b, 0,1,2,3,4,5,6,7,8,9,10,11,12,13,14,15);
}
DEV v16bf ld_frag_b(const unsigned short* p) {
  v8bf a = *(const v8bf*)p;
  v8bf b = *(const v8bf*)(p + 8);
  return __builtin_shufflevector(a, b, 0,1,2,3,4,5,6,7,8,9,10,11,12,13,14,15);
}
DEV v8f wmma_bf16(v16bf a, v16bf b, v8f c) {
  return __builtin_amdgcn_wmma_f32_16x16x32_bf16(false, a, false, b,
                                                 (short)0, c, false, false);
}

// ---------------------------------------------------------------------------
// Prep kernels
// ---------------------------------------------------------------------------
__global__ void k_split_x(const float* __restrict__ x,
                          unsigned short* __restrict__ xh,
                          unsigned short* __restrict__ xl, int n) {
  for (int i = blockIdx.x * blockDim.x + threadIdx.x; i < n;
       i += gridDim.x * blockDim.x) {
    float v = x[i];
    unsigned short h = f2bf_rne(v);
    xh[i] = h;
    xl[i] = f2bf_rne(v - bf2f(h));
  }
}

// w[R][C] -> hi/lo[C][R]   (transpose so B fragments are contiguous per lane)
__global__ void k_transpose_split(const float* __restrict__ w, int R, int C,
                                  unsigned short* __restrict__ hi,
                                  unsigned short* __restrict__ lo) {
  int n = R * C;
  for (int i = blockIdx.x * blockDim.x + threadIdx.x; i < n;
       i += gridDim.x * blockDim.x) {
    int r = i / C, c = i % C;
    float v = w[i];
    unsigned short h = f2bf_rne(v);
    hi[(size_t)c * R + r] = h;
    lo[(size_t)c * R + r] = f2bf_rne(v - bf2f(h));
  }
}

__global__ void k_init(unsigned short* __restrict__ sbuf,
                       unsigned* __restrict__ ctr) {
  int i = blockIdx.x * blockDim.x + threadIdx.x;
  for (int j = i; j < 2 * Bz * Hz; j += gridDim.x * blockDim.x) sbuf[j] = 0;
  if (i == 0) *ctr = 0;
}

// ---------------------------------------------------------------------------
// Phase A: input currents, stored DIRECTLY in the scan's fragment layout:
//   icf[t][tile][lane][v],  tile = (n>>1)*4 + (n&1)*2 + m,
//   element (b,h): m=b>>4, n=h>>4, lane=((b&15)>>3)*16 + (h&15), v=b&7.
// 3-term bf16 split ~ f32 accuracy; one 16x16 tile per wave; K = 256.
// ---------------------------------------------------------------------------
__global__ __launch_bounds__(256) void k_gemm_in(
    const unsigned short* __restrict__ xh, const unsigned short* __restrict__ xl,
    const unsigned short* __restrict__ wh, const unsigned short* __restrict__ wl,
    float* __restrict__ icf) {
  const int lane = threadIdx.x & 31;
  const int wave = threadIdx.x >> 5;
  const int tile = blockIdx.x * 8 + wave;
  const int mt = tile >> 5;              // 2048 M-tiles (rows r = b*1024+s)
  const int nt = tile & 31;              // 32 N-tiles (H/16)
  const int l16 = lane & 15;
  const int sel = lane >> 4;

  const unsigned short* pah = xh + (size_t)(mt * 16 + l16) * Iz + sel * 16;
  const unsigned short* pal = xl + (size_t)(mt * 16 + l16) * Iz + sel * 16;
  const unsigned short* pbh = wh + (size_t)(nt * 16 + l16) * Iz + sel * 16;
  const unsigned short* pbl = wl + (size_t)(nt * 16 + l16) * Iz + sel * 16;

  v8f c = {0.f, 0.f, 0.f, 0.f, 0.f, 0.f, 0.f, 0.f};
#pragma unroll
  for (int kb = 0; kb < Iz / 32; ++kb) {
    v16bf ah = ld_frag_a(pah + kb * 32);
    v16bf al = ld_frag_a(pal + kb * 32);
    v16bf bh = ld_frag_b(pbh + kb * 32);
    v16bf bl = ld_frag_b(pbl + kb * 32);
    c = wmma_bf16(ah, bh, c);
    c = wmma_bf16(al, bh, c);
    c = wmma_bf16(ah, bl, c);
  }
  // whole tile shares one batch index b (16 rows = 16 consecutive s)
  const int bb = mt >> 6;                        // b = (mt*16)>>10
  const int stile = (nt >> 1) * 4 + (nt & 1) * 2 + (bb >> 4);
  const int lane2 = ((bb & 15) >> 3) * 16 + l16;
  const int v2 = bb & 7;
  float* dbase = icf + (size_t)(stile * 256 + lane2 * 8 + v2);
#pragma unroll
  for (int v = 0; v < 8; ++v) {
    int s = (mt * 16 + v + 8 * sel) & 1023;      // C layout: M = v + 8*sel
    dbase[(size_t)s * (Bz * Hz)] = c[v];
  }
}

// ---------------------------------------------------------------------------
// Phase B: persistent scan. 16 WGs x 128 threads; wave owns one 16x16 tile
// of the [32 x 512] state for all 1024 steps (mp/refrac live in VGPRs).
// W_lat slice (bf16 hi+lo, n-major) staged into LDS via the Tensor Data
// Mover (padded rows via TDM pad_enable). Spikes ping-pong in L2.
// ---------------------------------------------------------------------------
static constexpr int ROWE = 520;  // LDS row stride in ushorts (1040 B, padded)

#ifdef HAVE_TDM
// 2-D TDM load: 32 rows x 512 bf16, row stride 512 in memory; LDS gets a
// 16-byte pad after every 1024 bytes (pad_interval=7 -> 256 DW, amount=3 -> 4 DW)
DEV void tdm_load_wlat_slice(const unsigned short* gsrc, unsigned lds_off) {
  unsigned long long ga = (unsigned long long)(uintptr_t)gsrc;
  v4u g0;
  g0[0] = 1u;                                          // count=1, user mode
  g0[1] = lds_off;                                     // lds_addr
  g0[2] = (unsigned)(ga & 0xFFFFFFFFu);                // global_addr[31:0]
  g0[3] = (unsigned)((ga >> 32) & 0x01FFFFFFu) | (2u << 30);  // type=2 "image"
  v8i g1;
  g1[0] = (int)((1u << 16) |          // data_size = 1 -> 2 bytes
                (1u << 20) |          // pad_enable
                (7u << 22) |          // pad_interval: 256 DWORDs = 1024 B
                (3u << 25));          // pad_amount:   4 DWORDs  = 16 B
  g1[1] = (int)(512u << 16);          // tensor_dim0[15:0] in [31:16]
  g1[2] = (int)(32u << 16);           // tensor_dim0 hi=0 ; tensor_dim1 lo=32
  g1[3] = (int)(512u << 16);          // tensor_dim1 hi=0 ; tile_dim0=512
  g1[4] = 32;                         // tile_dim1=32 ; tile_dim2=0
  g1[5] = 512;                        // tensor_dim0_stride lo32 = 512 elems
  g1[6] = 0;                          // stride hi ; tensor_dim1_stride lo
  g1[7] = 0;
  v4i z4 = {0, 0, 0, 0};
#if __clang_major__ >= 23
  v8i z8 = {0, 0, 0, 0, 0, 0, 0, 0};
  __builtin_amdgcn_tensor_load_to_lds(g0, g1, z4, z4, z8, 0);
#else
  __builtin_amdgcn_tensor_load_to_lds(g0, g1, z4, z4, 0);
#endif
}
#endif

__global__ __launch_bounds__(128) void k_scan(
    const float* __restrict__ icf,              // [S][64 tiles][256] fragments
    const unsigned short* __restrict__ wlat_h,  // [H][H] n-major
    const unsigned short* __restrict__ wlat_l,
    const float* __restrict__ thr,              // [H]
    unsigned short* __restrict__ sbuf,          // [2][B][H] bf16 ping-pong
    unsigned short* __restrict__ seq,           // [S][B][H] bf16
    unsigned* __restrict__ ctr) {
  extern __shared__ unsigned char smem[];
  unsigned short* lds_h = (unsigned short*)smem;
  unsigned short* lds_l = lds_h + 32 * ROWE;

  const int g = blockIdx.x;
  const int tid = threadIdx.x;
  const int lane = tid & 31;
  const int wave = tid >> 5;
  const int l16 = lane & 15;
  const int sel = lane >> 4;
  const int m = wave & 1;        // batch-tile (rows 0-15 / 16-31)
  const int ntl = wave >> 1;     // local n-tile (0/1)

  // ---- stage this WG's W_lat column slice into LDS ----
#ifdef HAVE_TDM
  if (wave == 0) {
    tdm_load_wlat_slice(wlat_h + (size_t)g * 32 * Hz,
                        (unsigned)(uintptr_t)lds_h);
    tdm_load_wlat_slice(wlat_l + (size_t)g * 32 * Hz,
                        (unsigned)(uintptr_t)lds_l);
    __builtin_amdgcn_s_wait_tensorcnt(0);
  }
#else
  for (int idx = tid; idx < 32 * Hz; idx += 128) {
    int nl = idx >> 9, k = idx & 511;
    size_t gsrc = (size_t)(g * 32 + nl) * Hz + k;
    lds_h[nl * ROWE + k] = wlat_h[gsrc];
    lds_l[nl * ROWE + k] = wlat_l[gsrc];
  }
#endif
  __syncthreads();

  const int nloc = ntl * 16 + l16;
  const int h = g * 32 + nloc;                  // global hidden index
  const float th = thr[h];
  const int arow = m * 16 + l16;                // spike row for A fragment
  const unsigned short* pbh = lds_h + nloc * ROWE + sel * 16;
  const unsigned short* pbl = lds_l + nloc * ROWE + sel * 16;
  const float* icp = icf + (size_t)(g * 4 + wave) * 256 + lane * 8;

  v8f mp = {0.f, 0.f, 0.f, 0.f, 0.f, 0.f, 0.f, 0.f};
  v8f rf = {0.f, 0.f, 0.f, 0.f, 0.f, 0.f, 0.f, 0.f};

  for (int t = 0; t < Sz; ++t) {
    // input currents for this tile: one contiguous 32B vector per lane;
    // issue early so it completes under the WMMAs. Prefetch next step.
    const v8f icv = *(const v8f*)(icp + (size_t)t * (Bz * Hz));
    __builtin_prefetch(icp + (size_t)(t + 1) * (Bz * Hz), 0, 0);

    const unsigned short* prev =
        sbuf + ((size_t)(t & 1)) * Bz * Hz + (size_t)arow * Hz + sel * 16;
    unsigned short* cur = sbuf + ((size_t)((t + 1) & 1)) * Bz * Hz;

    v8f c = {0.f, 0.f, 0.f, 0.f, 0.f, 0.f, 0.f, 0.f};
#pragma unroll
    for (int kb = 0; kb < Hz / 32; ++kb) {
      v16bf a = ld_frag_a(prev + kb * 32);
      v16bf bh = ld_frag_b(pbh + kb * 32);
      v16bf bl = ld_frag_b(pbl + kb * 32);
      c = wmma_bf16(a, bh, c);   // lateral (hi part)
      c = wmma_bf16(a, bl, c);   // + residual -> ~f32 accuracy
    }

#pragma unroll
    for (int v = 0; v < 8; ++v) {
      int b = m * 16 + v + 8 * sel;
      float m2 = DECAY * mp[v] + icv[v] - c[v];
      m2 = (rf[v] > 0.f) ? 0.f : m2;
      float r2 = fmaxf(rf[v] - 1.f, 0.f);
      float sp = (m2 >= th) ? 1.f : 0.f;
      mp[v] = (sp > 0.f) ? 0.f : m2;
      rf[v] = (sp > 0.f) ? REFRAC : r2;
      unsigned short spb = (sp > 0.f) ? (unsigned short)0x3F80 : (unsigned short)0;
      cur[(size_t)b * Hz + h] = spb;
      seq[((size_t)t * Bz + b) * Hz + h] = spb;
    }

    // device-wide barrier: monotonic counter, target NWG*(t+1)
    __threadfence();
    __syncthreads();
    if (tid == 0) {
      __hip_atomic_fetch_add(ctr, 1u, __ATOMIC_ACQ_REL, __HIP_MEMORY_SCOPE_AGENT);
      unsigned target = (unsigned)(NWG * (t + 1));
      while (__hip_atomic_load(ctr, __ATOMIC_ACQUIRE, __HIP_MEMORY_SCOPE_AGENT) <
             target) {
        __builtin_amdgcn_s_sleep(1);
      }
    }
    __syncthreads();
    __threadfence();
  }
}

// ---------------------------------------------------------------------------
// Phase C: out[b][s][o] = spikes_seq[s][b][:] @ W_out  (spikes exact in bf16)
// ---------------------------------------------------------------------------
__global__ __launch_bounds__(256) void k_gemm_out(
    const unsigned short* __restrict__ sq,       // [32768][512], row = s*32+b
    const unsigned short* __restrict__ wh,       // [O][H] n-major
    const unsigned short* __restrict__ wl,
    float* __restrict__ outp) {                  // [B][S][O]
  const int lane = threadIdx.x & 31;
  const int wave = threadIdx.x >> 5;
  const int tile = blockIdx.x * 8 + wave;
  const int mt = tile >> 4;              // 2048 M-tiles
  const int nt = tile & 15;              // 16 N-tiles (O/16)
  const int l16 = lane & 15;
  const int sel = lane >> 4;

  const unsigned short* pa = sq + (size_t)(mt * 16 + l16) * Hz + sel * 16;
  const unsigned short* pbh = wh + (size_t)(nt * 16 + l16) * Hz + sel * 16;
  const unsigned short* pbl = wl + (size_t)(nt * 16 + l16) * Hz + sel * 16;

  v8f c = {0.f, 0.f, 0.f, 0.f, 0.f, 0.f, 0.f, 0.f};
#pragma unroll
  for (int kb = 0; kb < Hz / 32; ++kb) {
    v16bf a = ld_frag_a(pa + kb * 32);
    v16bf bh = ld_frag_b(pbh + kb * 32);
    v16bf bl = ld_frag_b(pbl + kb * 32);
    c = wmma_bf16(a, bh, c);
    c = wmma_bf16(a, bl, c);
  }
  const int o = nt * 16 + l16;
#pragma unroll
  for (int v = 0; v < 8; ++v) {
    int r = mt * 16 + v + 8 * sel;
    int s = r >> 5, b = r & 31;          // row = s*32 + b
    outp[(((size_t)b << 10) + s) * Oz + o] = c[v];
  }
}

// ---------------------------------------------------------------------------
// Launch
// ---------------------------------------------------------------------------
extern "C" void kernel_launch(void* const* d_in, const int* in_sizes, int n_in,
                              void* d_out, int out_size, void* d_ws,
                              size_t ws_size, hipStream_t stream) {
  (void)in_sizes; (void)n_in; (void)out_size;
  const float* x = (const float*)d_in[0];
  const float* w_in = (const float*)d_in[1];
  const float* w_lat = (const float*)d_in[2];
  const float* w_out = (const float*)d_in[3];
  const float* thr = (const float*)d_in[4];
  float* outp = (float*)d_out;

  uint8_t* ws = (uint8_t*)d_ws;
  size_t o_ic = 0;                                      // icf fragments f32
  size_t o_seq = o_ic + (size_t)Mz * Hz * 4;            // spikes_seq bf16
  size_t o_xh = o_seq + (size_t)Mz * Hz * 2;            // x hi bf16
  size_t o_xl = o_xh + (size_t)Mz * Iz * 2;             // x lo bf16
  size_t o_winh = o_xl + (size_t)Mz * Iz * 2;           // W_in^T hi
  size_t o_winl = o_winh + (size_t)Hz * Iz * 2;
  size_t o_wlh = o_winl + (size_t)Hz * Iz * 2;          // W_lat^T hi
  size_t o_wll = o_wlh + (size_t)Hz * Hz * 2;
  size_t o_woh = o_wll + (size_t)Hz * Hz * 2;           // W_out^T hi
  size_t o_wol = o_woh + (size_t)Oz * Hz * 2;
  size_t o_sb = o_wol + (size_t)Oz * Hz * 2;            // spike ping-pong
  size_t o_ctr = o_sb + 2 * (size_t)Bz * Hz * 2;
  size_t need = o_ctr + 256;
  if (ws_size < need) return;

  float* icf = (float*)(ws + o_ic);
  unsigned short* seq = (unsigned short*)(ws + o_seq);
  unsigned short* xh = (unsigned short*)(ws + o_xh);
  unsigned short* xl = (unsigned short*)(ws + o_xl);
  unsigned short* winh = (unsigned short*)(ws + o_winh);
  unsigned short* winl = (unsigned short*)(ws + o_winl);
  unsigned short* wlh = (unsigned short*)(ws + o_wlh);
  unsigned short* wll = (unsigned short*)(ws + o_wll);
  unsigned short* woh = (unsigned short*)(ws + o_woh);
  unsigned short* wol = (unsigned short*)(ws + o_wol);
  unsigned short* sb = (unsigned short*)(ws + o_sb);
  unsigned* ctr = (unsigned*)(ws + o_ctr);

  k_split_x<<<2048, 256, 0, stream>>>(x, xh, xl, Mz * Iz);
  k_transpose_split<<<512, 256, 0, stream>>>(w_in, Iz, Hz, winh, winl);
  k_transpose_split<<<1024, 256, 0, stream>>>(w_lat, Hz, Hz, wlh, wll);
  k_transpose_split<<<512, 256, 0, stream>>>(w_out, Hz, Oz, woh, wol);
  k_init<<<32, 256, 0, stream>>>(sb, ctr);

  k_gemm_in<<<(Mz / 16) * (Hz / 16) / 8, 256, 0, stream>>>(xh, xl, winh, winl,
                                                           icf);

  size_t scan_lds = (size_t)2 * 32 * ROWE * sizeof(unsigned short);  // 66,560 B
  k_scan<<<NWG, 128, scan_lds, stream>>>(icf, wlh, wll, thr, sb, seq, ctr);

  k_gemm_out<<<(Mz / 16) * (Oz / 16) / 8, 256, 0, stream>>>(seq, woh, wol, outp);
}
